// OneCardPokerRatPaynet_51685636440662
// MI455X (gfx1250) — matrix operand order repaired and checked by reading, not compiled, not run on hardware
//
#include <hip/hip_runtime.h>

#define QRE_ITERS  100
#define QRE_MINLAM 0.1f

typedef __attribute__((ext_vector_type(16))) _Float16 v16h;
typedef __attribute__((ext_vector_type(8)))  _Float16 v8h;
typedef __attribute__((ext_vector_type(8)))  float    v8f;

// One-card-poker payoff entry P[r][c] (already scaled by 1/(n(n-1)), n=64).
// Note P^T == -P, which lets both GEMMs of the fixed point share one LDS copy.
__device__ __forceinline__ float p_entry(int r, int c) {
    const float norm = 1.0f / (64.0f * 63.0f);
    int rt = r >> 7, ct = c >> 7;          // top-half flags (rows/cols >= 2n)
    int rp = r & 1,  cp = c & 1;           // parity
    int a = (r & 127) >> 1, b = (c & 127) >> 1;
    float sgn = (b > a) ? 1.0f : ((b < a) ? -1.0f : 0.0f);
    float off = (a != b) ? 1.0f : 0.0f;
    float v = 0.0f;
    if      (!rt && !rp && !ct && !cp) v = sgn;         // init (i0,i0)
    else if ( rt && !rp && !ct &&  cp) v = off;         // init (i2,i1)
    else if (!rt &&  rp &&  ct && !cp) v = -off;        // init (i1,i2)
    else if ( rt &&  rp && !ct &&  cp) v = 2.0f * sgn;  // raise (i3,i1)
    else if (!rt &&  rp &&  ct &&  cp) v = 2.0f * sgn;  // raise (i1,i3)
    return v * norm;
}

// XOR swizzle on 16B chunks: rows are 512B apart (2 bank periods), so
// chunk_phys = chunk_log ^ (row & 15) spreads row-strided b128 loads
// across all 64 LDS banks. Scalar element index helper:
__device__ __forceinline__ int swz16(int r, int c) {
    return (((c >> 3) ^ (r & 15)) << 3) | (c & 7);
}

// One wave computes two 16x16 f32 output tiles of  [16 x 256] = A[16x256] * B,
// where B[k, n] = P[i0 + n, k]  (contiguous rows of LDS-resident P).
// A layout  (16x32):  lane m<16: K 0-15; lane 16+: K 16-31
// B layout  (contiguous K, same as A)  :  lanes 0-15: K 0-15; lanes 16-31: K 16-31
__device__ __forceinline__ void wave_gemm16(const _Float16* __restrict__ A,
                                            const _Float16* __restrict__ P,
                                            float* __restrict__ GH,
                                            int w, int lane) {
    const int nn  = lane & 15;
    const int sel = lane >> 4;
    const v8h* Ach = (const v8h*)A;
    const v8h* Pch = (const v8h*)P;
    const int m  = nn;             // A row owned by this lane
    const int r0 = w * 32 + nn;    // P row feeding tile-0 column nn
    const int r1 = r0 + 16;        // P row feeding tile-1 column nn
    v8f acc0 = {0.f, 0.f, 0.f, 0.f, 0.f, 0.f, 0.f, 0.f};
    v8f acc1 = acc0;
#pragma unroll
    for (int kk = 0; kk < 8; ++kk) {       // K = 256 in steps of 32
        const int kc = kk * 4;             // 16B-chunk base of this K slice
        v8h alo = Ach[m * 32 + ((kc + 2 * sel)     ^ m)];
        v8h ahi = Ach[m * 32 + ((kc + 2 * sel + 1) ^ m)];
        v16h a = __builtin_shufflevector(alo, ahi,
                 0,1,2,3,4,5,6,7,8,9,10,11,12,13,14,15);
        v8h b0l = Pch[r0 * 32 + ((kc + 2 * sel)     ^ nn)];
        v8h b0h = Pch[r0 * 32 + ((kc + 2 * sel + 1) ^ nn)];
        v16h b0 = __builtin_shufflevector(b0l, b0h,
                  0,1,2,3,4,5,6,7,8,9,10,11,12,13,14,15);
        acc0 = __builtin_amdgcn_wmma_f32_16x16x32_f16(
                   false, a, false, b0, (short)0, acc0, false, false);
        v8h b1l = Pch[r1 * 32 + ((kc + 2 * sel)     ^ nn)];
        v8h b1h = Pch[r1 * 32 + ((kc + 2 * sel + 1) ^ nn)];
        v16h b1 = __builtin_shufflevector(b1l, b1h,
                  0,1,2,3,4,5,6,7,8,9,10,11,12,13,14,15);
        acc1 = __builtin_amdgcn_wmma_f32_16x16x32_f16(
                   false, a, false, b1, (short)0, acc1, false, false);
    }
    // C/D layout: lane = column (0-15 / 16-31), VGPR r = row r (+8 for hi half)
    const int i0 = w * 32;
#pragma unroll
    for (int r = 0; r < 8; ++r) {
        const int mrow = r + sel * 8;
        GH[mrow * 256 + i0 + nn]      = acc0[r];
        GH[mrow * 256 + i0 + 16 + nn] = acc1[r];
    }
}

extern "C" __global__ void __launch_bounds__(256)
qre_poker_kernel(const float* __restrict__ x, const float* __restrict__ W,
                 float* __restrict__ out_u, float* __restrict__ out_v) {
    __shared__ _Float16 Pl[256 * 256];   // 128 KB payoff matrix (swizzled rows)
    __shared__ _Float16 uh[16 * 256];    //   8 KB u state, f16 (GEMM A operand)
    __shared__ _Float16 vh[16 * 256];    //   8 KB v state, f16
    __shared__ float    gh[16 * 256];    //  16 KB g/h GEMM output (f32)
    // total: 160 KB — one workgroup per WGP

    const int t    = threadIdx.x;
    const int wg   = blockIdx.x;
    const int lane = t & 31;
    const int w    = t >> 5;

    // ---- per-batch-row lambdas kept in registers: lam_c = x[b]*W[c] + 0.1 ----
    const float w0 = W[0], w1 = W[1], w2 = W[2], w3 = W[3];
    float xu[4], xv[8];
#pragma unroll
    for (int i = 0; i < 4; ++i) xu[i] = x[wg * 16 + ((t >> 6) + 4 * i)];
#pragma unroll
    for (int i = 0; i < 8; ++i) xv[i] = x[wg * 16 + ((t >> 7) + 2 * i)];

    // ---- build P in LDS (f16, swizzled); thread t owns row t ----
    {
        const int r = t, rx = r & 15;
        for (int c8 = 0; c8 < 32; ++c8) {
            const int pc = (c8 ^ rx) << 3;
#pragma unroll
            for (int e = 0; e < 8; ++e)
                Pl[r * 256 + pc + e] = (_Float16)p_entry(r, c8 * 8 + e);
        }
    }
    // ---- initial state: u0 = [0.5 x128 | 0.25 x128], v0 = 0.5 ----
    for (int i = t; i < 16 * 256; i += 256) {
        const int b = i >> 8, c = i & 255;
        const int si = b * 256 + swz16(b, c);
        uh[si] = (_Float16)((c < 128) ? 0.5f : 0.25f);
        vh[si] = (_Float16)0.5f;
    }
    __syncthreads();

    for (int iter = 0; iter < QRE_ITERS; ++iter) {
        const bool last = (iter == QRE_ITERS - 1);

        // ---- g = v @ P^T   (B[k,n] = P[i0+n, k]) ----
        wave_gemm16(vh, Pl, gh, w, lane);
        __syncthreads();

        // ---- u <- smoothed best response (min player, 2-level tree) ----
#pragma unroll
        for (int i = 0; i < 4; ++i) {
            const int idx = t + i * 256;
            const int b = idx >> 6, kk = idx & 63;
            const float l0 = fmaf(xu[i], w0, QRE_MINLAM);
            const float l1 = fmaf(xu[i], w1, QRE_MINLAM);
            const float* gb = gh + b * 256;
            const float g0r = gb[2 * kk],       g1r = gb[2 * kk + 1];
            const float g0l = gb[128 + 2 * kk], g1l = gb[129 + 2 * kk];
            // leaf infoset: J = -l1*logsumexp(-g/l1), b_leaf = softmax(-g/l1)
            const float a0 = -g0l / l1, a1 = -g1l / l1;
            const float mx = fmaxf(a0, a1);
            const float e0 = __expf(a0 - mx), e1 = __expf(a1 - mx);
            const float s  = e0 + e1;
            const float J  = -l1 * (mx + __logf(s));
            const float bl0 = e0 / s, bl1 = e1 / s;
            // root infoset: b_root = softmax(-(g + [J,0]) / l0)
            const float c0 = -(g0r + J) / l0, c1 = -g1r / l0;
            const float mr = fmaxf(c0, c1);
            const float f0 = __expf(c0 - mr), f1 = __expf(c1 - mr);
            const float sr = f0 + f1;
            const float br0 = f0 / sr, br1 = f1 / sr;
            const float ul0 = br0 * bl0, ul1 = br0 * bl1;
            _Float16* ub = uh + b * 256;
            ub[swz16(b, 2 * kk)]       = (_Float16)br0;
            ub[swz16(b, 2 * kk + 1)]   = (_Float16)br1;
            ub[swz16(b, 128 + 2 * kk)] = (_Float16)ul0;
            ub[swz16(b, 129 + 2 * kk)] = (_Float16)ul1;
            if (last) {
                float* ob = out_u + (size_t)(wg * 16 + b) * 256;
                ob[2 * kk] = br0;       ob[2 * kk + 1] = br1;
                ob[128 + 2 * kk] = ul0; ob[129 + 2 * kk] = ul1;
            }
        }
        __syncthreads();

        // ---- h-pass: acc[m,n] = sum_i u[i] * P[j0+n, i] = -(u@P)  (P^T = -P) ----
        wave_gemm16(uh, Pl, gh, w, lane);
        __syncthreads();

        // ---- v <- softmax(h / lv) pairwise, with h = -acc ----
#pragma unroll
        for (int i = 0; i < 8; ++i) {
            const int idx = t + i * 256;
            const int b = idx >> 7, k = idx & 127;
            const float lv = fmaf(xv[i], (k < 64) ? w2 : w3, QRE_MINLAM);
            const float p0 = gh[b * 256 + 2 * k], p1 = gh[b * 256 + 2 * k + 1];
            const float h0 = -p0 / lv, h1 = -p1 / lv;
            const float mx = fmaxf(h0, h1);
            const float e0 = __expf(h0 - mx), e1 = __expf(h1 - mx);
            const float s  = e0 + e1;
            const float vv0 = e0 / s, vv1 = e1 / s;
            _Float16* vb = vh + b * 256;
            vb[swz16(b, 2 * k)]     = (_Float16)vv0;
            vb[swz16(b, 2 * k + 1)] = (_Float16)vv1;
            if (last) {
                float* ob = out_v + (size_t)(wg * 16 + b) * 256;
                ob[2 * k] = vv0; ob[2 * k + 1] = vv1;
            }
        }
        __syncthreads();
    }
}

extern "C" void kernel_launch(void* const* d_in, const int* in_sizes, int n_in,
                              void* d_out, int out_size, void* d_ws, size_t ws_size,
                              hipStream_t stream) {
    const float* x = (const float*)d_in[0];   // [1024, 1] f32
    const float* W = (const float*)d_in[1];   // [4, 1]    f32
    float* out   = (float*)d_out;             // u[1024*256] then v[1024*256]
    float* out_u = out;
    float* out_v = out + 1024 * 256;
    qre_poker_kernel<<<64, 256, 0, stream>>>(x, W, out_u, out_v);
}
